// Sequence_Attention_Layer_49520972923438
// MI455X (gfx1250) — matrix operand order repaired and checked by reading, not compiled, not run on hardware
//
#include <hip/hip_runtime.h>

typedef __attribute__((ext_vector_type(16))) __bf16 v16bf;
typedef __attribute__((ext_vector_type(8)))  float  v8f;
typedef __attribute__((ext_vector_type(4)))  float  f4;

#define BATCH 32
#define N_SEQ 2048
#define U_DIM 1024
#define DA    512
#define ROWS_PER_BLOCK 64
#define KT_COUNT (U_DIM / 32)   // 32 k-tiles of 32
#define DT_COUNT (DA / 16)      // 32 d-tiles of 16
// Fragment-major LDS: frag (kt, mt) occupies 256 dwords; lane's 8 dwords contiguous.
#define LDS_DW   (KT_COUNT * 4 * 256)       // 32768 dwords
#define LDS_BYTES (LDS_DW * 4)              // 131072 B -> 2 blocks fit in 320 KB WGP LDS

union FragBF {
    unsigned int u[8];
    uint4 q[2];
    v16bf v;
};

__device__ __forceinline__ unsigned int pack_bf16(float a, float b) {
    unsigned int ua = __builtin_bit_cast(unsigned int, a);
    unsigned int ub = __builtin_bit_cast(unsigned int, b);
    ua = (ua + 0x7FFFu + ((ua >> 16) & 1u)) >> 16;   // round-to-nearest-even
    ub = (ub + 0x7FFFu + ((ub >> 16) & 1u)) >> 16;
    return (ua & 0xFFFFu) | (ub << 16);
}

// ---------------------------------------------------------------------------
// W (f32 [1024,512]) -> bf16 fragment-major: frag (dt,kt) is 256 dwords,
// dword index = ((dt*KT_COUNT + kt)*256 + lane*8 + j).
// B 32x16 bf16 frag: col N = lane%16, K = kt*32 + (lane/16)*16 + 2j (+1 hi).
// ---------------------------------------------------------------------------
__global__ void repack_w_kernel(const float* __restrict__ W,
                                unsigned int* __restrict__ Wp) {
    int idx = blockIdx.x * blockDim.x + threadIdx.x;
    if (idx >= DT_COUNT * KT_COUNT * 256) return;
    int j    = idx & 7;
    int lane = (idx >> 3) & 31;
    int kt   = (idx >> 8) & 31;
    int dt   = idx >> 13;
    int col  = dt * 16 + (lane & 15);
    int k0   = kt * 32 + (lane >> 4) * 16 + 2 * j;
    float a = W[(size_t)k0 * DA + col];
    float b = W[(size_t)(k0 + 1) * DA + col];
    Wp[idx] = pack_bf16(a, b);
}

__global__ void zero_scores_kernel(float* __restrict__ sc) {
    int i = blockIdx.x * blockDim.x + threadIdx.x;
    if (i < BATCH * N_SEQ) sc[i] = 0.0f;
}

// ---------------------------------------------------------------------------
// scores[row] = sum_d tanh( (H W)[row,d] + B[d] ) * U[d]
// 256 thr / 8 waves, 64 rows. Fragment-major bf16 A in LDS, each wave runs
// two d-tiles per pass so every A fragment read feeds two WMMAs.
// ---------------------------------------------------------------------------
__global__ __launch_bounds__(256)
void score_kernel(const float* __restrict__ H,
                  const float* __restrict__ Bb,
                  const float* __restrict__ Uv,
                  const unsigned int* __restrict__ Wp,
                  float* __restrict__ scores) {
    extern __shared__ unsigned int ldsA[];   // fragment-major, 128 KB
    const int tid  = threadIdx.x;
    const int row0 = blockIdx.x * ROWS_PER_BLOCK;

    // Stage 64 rows of H as bf16 A-fragments (coalesced float2 global reads,
    // scattered but conflict-light LDS writes).
    const float* hblk = H + (size_t)row0 * U_DIM;
    for (int idx = tid; idx < ROWS_PER_BLOCK * (U_DIM / 2); idx += 256) {
        int r  = idx >> 9;              // row in block (0..63)
        int kd = idx & 511;             // dword along k (pairs of bf16)
        float2 f = ((const float2*)(hblk + (size_t)r * U_DIM))[kd];
        int mt    = r >> 4;
        int lmod  = r & 15;
        int k     = kd << 1;
        int kt    = k >> 5;
        int krem  = k & 31;             // k = lhalf*16 + 2*j
        int lhalf = krem >> 4;
        int j     = (krem >> 1) & 7;
        int ln    = lmod | (lhalf << 4);
        ldsA[(((kt << 2) + mt) << 8) + (ln << 3) + j] = pack_bf16(f.x, f.y);
    }
    __syncthreads();

    const int wave  = tid >> 5;     // 0..7
    const int lane  = tid & 31;
    const int lhalf = lane >> 4;
    const int lmod  = lane & 15;

    float psum[32];                 // [mt][r] partial row sums over this wave's d
#pragma unroll
    for (int i = 0; i < 32; ++i) psum[i] = 0.0f;

    const unsigned int* aBase = ldsA + (lane << 3);   // + frag*256 (imm/uniform)

#pragma unroll
    for (int di = 0; di < 2; ++di) {
        const int dt0  = (wave << 2) + (di << 1);     // wave owns dt [4w, 4w+4)
        const int dc0  = dt0 * 16 + lmod;
        const int dc1  = dc0 + 16;
        const float b0 = Bb[dc0], b1 = Bb[dc1];
        const float u0 = Uv[dc0], u1 = Uv[dc1];

        v8f acc0[4], acc1[4];
#pragma unroll
        for (int mt = 0; mt < 4; ++mt)
#pragma unroll
            for (int r = 0; r < 8; ++r) { acc0[mt][r] = 0.0f; acc1[mt][r] = 0.0f; }

        const unsigned int* wp0 = Wp + ((size_t)dt0 * KT_COUNT) * 256 + lane * 8;
        const unsigned int* wp1 = wp0 + (size_t)KT_COUNT * 256;

        for (int kt = 0; kt < KT_COUNT; ++kt) {
            FragBF bf0, bf1;
            const uint4* bp0 = (const uint4*)(wp0 + (size_t)kt * 256);
            const uint4* bp1 = (const uint4*)(wp1 + (size_t)kt * 256);
            bf0.q[0] = bp0[0]; bf0.q[1] = bp0[1];
            bf1.q[0] = bp1[0]; bf1.q[1] = bp1[1];
#pragma unroll
            for (int mt = 0; mt < 4; ++mt) {
                FragBF af;
                const unsigned int* ap = aBase + (((kt << 2) + mt) << 8);
                af.q[0] = *(const uint4*)ap;
                af.q[1] = *(const uint4*)(ap + 4);
                acc0[mt] = __builtin_amdgcn_wmma_f32_16x16x32_bf16(
                    false, af.v, false, bf0.v, (short)0, acc0[mt], false, false);
                acc1[mt] = __builtin_amdgcn_wmma_f32_16x16x32_bf16(
                    false, af.v, false, bf1.v, (short)0, acc1[mt], false, false);
            }
        }
        // epilogue: tanh + scale by U, fold both d-tiles into per-row partials
#pragma unroll
        for (int mt = 0; mt < 4; ++mt)
#pragma unroll
            for (int r = 0; r < 8; ++r)
                psum[mt * 8 + r] += tanhf(acc0[mt][r] + b0) * u0
                                  + tanhf(acc1[mt][r] + b1) * u1;
    }

    // Reduce over the 16 lanes of each half (each lane = one d column).
#pragma unroll
    for (int off = 1; off <= 8; off <<= 1)
#pragma unroll
        for (int i = 0; i < 32; ++i)
            psum[i] += __shfl_xor(psum[i], off, 32);

    if (lmod == 0) {
        const int rbase = row0 + lhalf * 8;   // lanes 0-15: M 0..7, 16-31: M 8..15
#pragma unroll
        for (int mt = 0; mt < 4; ++mt)
#pragma unroll
            for (int r = 0; r < 8; ++r)
                atomicAdd(&scores[rbase + mt * 16 + r], psum[mt * 8 + r]);
    }
}

// ---------------------------------------------------------------------------
// In-place softmax over the 2048-long sequence, one block per batch.
// ---------------------------------------------------------------------------
__global__ void softmax_kernel(float* __restrict__ sc) {
    __shared__ float red[8];
    float* s = sc + (size_t)blockIdx.x * N_SEQ;
    const int tid = threadIdx.x, wave = tid >> 5, lane = tid & 31;

    float m = -3.402823466e38f;
    for (int i = tid; i < N_SEQ; i += 256) m = fmaxf(m, s[i]);
#pragma unroll
    for (int off = 1; off <= 16; off <<= 1) m = fmaxf(m, __shfl_xor(m, off, 32));
    if (lane == 0) red[wave] = m;
    __syncthreads();
    float M = red[0];
#pragma unroll
    for (int w = 1; w < 8; ++w) M = fmaxf(M, red[w]);
    __syncthreads();

    float sum = 0.0f;
    for (int i = tid; i < N_SEQ; i += 256) {
        float e = __expf(s[i] - M);
        s[i] = e;
        sum += e;
    }
#pragma unroll
    for (int off = 1; off <= 16; off <<= 1) sum += __shfl_xor(sum, off, 32);
    if (lane == 0) red[wave] = sum;
    __syncthreads();
    float S = 0.0f;
#pragma unroll
    for (int w = 0; w < 8; ++w) S += red[w];
    const float inv = 1.0f / S;
    for (int i = tid; i < N_SEQ; i += 256) s[i] *= inv;
}

// ---------------------------------------------------------------------------
// Ma = H * vu[row]; pure streaming, NT stores keep the 268 MB output out of L2.
// ---------------------------------------------------------------------------
__global__ void weight_kernel(const float* __restrict__ H,
                              const float* __restrict__ vu,
                              float* __restrict__ Ma) {
    const size_t total  = (size_t)BATCH * N_SEQ * (U_DIM / 4);
    const size_t stride = (size_t)gridDim.x * blockDim.x;
    const f4* h4 = (const f4*)H;
    f4* m4 = (f4*)Ma;
    for (size_t i = (size_t)blockIdx.x * blockDim.x + threadIdx.x; i < total; i += stride) {
        size_t row = i >> 8;                 // 256 f4 per 1024-wide row
        float w = vu[row];
        f4 h = h4[i];
        f4 o = h * w;
        __builtin_nontemporal_store(o, &m4[i]);
    }
}

extern "C" void kernel_launch(void* const* d_in, const int* in_sizes, int n_in,
                              void* d_out, int out_size, void* d_ws, size_t ws_size,
                              hipStream_t stream) {
    const float* H  = (const float*)d_in[0];
    const float* W  = (const float*)d_in[1];
    const float* Bb = (const float*)d_in[2];
    const float* Uv = (const float*)d_in[3];

    float* Ma = (float*)d_out;
    float* vu = Ma + (size_t)BATCH * N_SEQ * U_DIM;   // VU region of d_out
    unsigned int* Wp = (unsigned int*)d_ws;           // 1 MB bf16-repacked W

    repack_w_kernel<<<(DT_COUNT * KT_COUNT * 256) / 256, 256, 0, stream>>>(W, Wp);
    zero_scores_kernel<<<(BATCH * N_SEQ) / 256, 256, 0, stream>>>(vu);
    score_kernel<<<(BATCH * N_SEQ) / ROWS_PER_BLOCK, 256, LDS_BYTES, stream>>>(H, Bb, Uv, Wp, vu);
    softmax_kernel<<<BATCH, 256, 0, stream>>>(vu);
    weight_kernel<<<4096, 256, 0, stream>>>(H, vu, Ma);
}
